// ABCNN_22093311771258
// MI455X (gfx1250) — matrix operand order repaired and checked by reading, not compiled, not run on hardware
//
#include <hip/hip_runtime.h>
#include <hip/hip_bf16.h>

#define B_   256
#define L_   256
#define D_   100
#define KP_  128      // padded K for match-score GEMM (bf16)
#define NP_  112      // padded N (=D) for f-GEMM tiles (7 x 16)
#define LIN_ 512
#define FIN_ 600      // D*(1+NUM_LAYER)*2
#define NLAYER_ 2
#define EPS_ 1e-5f

typedef __bf16 bf16_t;
typedef bf16_t v16bf __attribute__((ext_vector_type(16)));
typedef bf16_t v8bf  __attribute__((ext_vector_type(8)));
typedef float  v8f   __attribute__((ext_vector_type(8)));
typedef float  v4f   __attribute__((ext_vector_type(4)));

// ---------------------------------------------------------------------------
// Embedding gather: e[b,l,d] = emb[q[b,l], d]; valid[b,l] = (q != 0)
// ---------------------------------------------------------------------------
__global__ void gather_kernel(const int* __restrict__ q,
                              const float* __restrict__ emb,
                              float* __restrict__ e,
                              float* __restrict__ valid) {
    int idx = blockIdx.x * blockDim.x + threadIdx.x;
    if (idx >= B_ * L_ * D_) return;
    int row = idx / D_;
    int d   = idx - row * D_;
    int tok = q[row];
    e[idx] = emb[(size_t)tok * D_ + d];
    if (d == 0) valid[row] = (tok != 0) ? 1.0f : 0.0f;
}

// ---------------------------------------------------------------------------
// Mean over L: dst_x[b, col0 + d] = mean_l src[b,l,d]   (x row stride FIN_)
// ---------------------------------------------------------------------------
__global__ void mean_kernel(const float* __restrict__ src,
                            float* __restrict__ x, int col0) {
    int idx = blockIdx.x * blockDim.x + threadIdx.x;
    if (idx >= B_ * D_) return;
    int b = idx / D_;
    int d = idx - b * D_;
    float s = 0.0f;
    const float* p = src + ((size_t)b * L_) * D_ + d;
    for (int l = 0; l < L_; ++l) s += p[(size_t)l * D_];
    x[(size_t)b * FIN_ + col0 + d] = s * (1.0f / (float)L_);
}

// ---------------------------------------------------------------------------
// Pack masked rows to bf16 (K padded with zeros) + squared-norm per row.
// One 128-thread block per (b,l) row; LDS tree reduction for sq.
// ---------------------------------------------------------------------------
__global__ __launch_bounds__(128)
void pack_kernel(const float* __restrict__ src, const float* __restrict__ valid,
                 bf16_t* __restrict__ sb, float* __restrict__ sq) {
    int row = blockIdx.x;          // b*L + l
    int t   = threadIdx.x;         // 0..127
    float v = 0.0f;
    if (t < D_) v = src[(size_t)row * D_ + t] * valid[row];
    sb[(size_t)row * KP_ + t] = (bf16_t)v;
    __shared__ float red[128];
    red[t] = v * v;
    __syncthreads();
    for (int s = 64; s > 0; s >>= 1) {
        if (t < s) red[t] += red[t + s];
        __syncthreads();
    }
    if (t == 0) sq[row] = red[0];
}

// ---------------------------------------------------------------------------
// Match-score GEMM: cross = s1 . s2^T via v_wmma_f32_16x16x32_bf16, fused
// epilogue A = 1/(1 + sqrt(max(sq1_i + sq2_j - 2*cross, 0))).
// Register blocked: one wave computes a 16x64 strip (1 A-frag x 4 B-frags),
// 16 wmma per wave. grid = (L/16, L/64, B), block = 32.
// ---------------------------------------------------------------------------
__global__ __launch_bounds__(32)
void msgemm_kernel(const bf16_t* __restrict__ sb1, const bf16_t* __restrict__ sb2,
                   const float* __restrict__ sq1, const float* __restrict__ sq2,
                   float* __restrict__ A) {
    const int lane = threadIdx.x;
    const int h  = lane >> 4;      // K-half selector
    const int ln = lane & 15;      // M (A-frag) / N (B-frag, C)
    const int ti  = blockIdx.x;
    const int tj0 = blockIdx.y * 4;
    const int b   = blockIdx.z;

    const bf16_t* arow = sb1 + ((size_t)b * L_ + ti * 16 + ln) * KP_;
    const bf16_t* brow = sb2 + ((size_t)b * L_ + tj0 * 16 + ln) * KP_;

    v8f c[4] = {};
#pragma unroll
    for (int kc = 0; kc < KP_; kc += 32) {
        v8bf alo = *(const v8bf*)(arow + kc + 8 * h);
        v8bf ahi = *(const v8bf*)(arow + kc + 16 + 8 * h);
        v16bf a;
#pragma unroll
        for (int t = 0; t < 8; ++t) { a[t] = alo[t]; a[t + 8] = ahi[t]; }
#pragma unroll
        for (int jt = 0; jt < 4; ++jt) {
            const bf16_t* br = brow + (size_t)jt * 16 * KP_;
            v8bf blo = *(const v8bf*)(br + kc + 8 * h);
            v8bf bhi = *(const v8bf*)(br + kc + 16 + 8 * h);
            v16bf bb;
#pragma unroll
            for (int t = 0; t < 8; ++t) { bb[t] = blo[t]; bb[t + 8] = bhi[t]; }
            c[jt] = __builtin_amdgcn_wmma_f32_16x16x32_bf16(false, a, false, bb,
                                                            (short)0, c[jt], false, false);
        }
    }

    float* Ab = A + (size_t)b * L_ * L_;
#pragma unroll
    for (int jt = 0; jt < 4; ++jt) {
        const float q2v = sq2[(size_t)b * L_ + (tj0 + jt) * 16 + ln];
#pragma unroll
        for (int r = 0; r < 8; ++r) {
            int i = ti * 16 + r + 8 * h;            // C layout: M = r + 8h
            float d2 = sq1[(size_t)b * L_ + i] + q2v - 2.0f * c[jt][r];
            d2 = fmaxf(d2, 0.0f);
            Ab[(size_t)i * L_ + (tj0 + jt) * 16 + ln] = 1.0f / (1.0f + sqrtf(d2));
        }
    }
}

// ---------------------------------------------------------------------------
// Attention GEMM: F = A @ W (transA=0) or A^T @ W (transA=1).
// A is fp32 [L,L] converted to bf16 on the fly; Wt is bf16 [NP_, L] (W^T,
// zero padded) so B-fragment loads are contiguous. Register blocked: one
// wave computes a 64x16 strip (4 A-frags x 1 shared B-frag), 32 wmma per
// wave. grid = (L/64, NP/16, B), block = 32.
// ---------------------------------------------------------------------------
__global__ __launch_bounds__(32)
void fgemm_kernel(const float* __restrict__ A, const bf16_t* __restrict__ Wt,
                  float* __restrict__ F, int transA) {
    const int lane = threadIdx.x;
    const int h  = lane >> 4;
    const int ln = lane & 15;
    const int ti0 = blockIdx.x * 4;
    const int tn  = blockIdx.y;
    const int b   = blockIdx.z;

    const float* Ab = A + (size_t)b * L_ * L_;
    const bf16_t* wrow = Wt + (size_t)(tn * 16 + ln) * L_;

    v8f c[4] = {};
#pragma unroll 2
    for (int kc = 0; kc < L_; kc += 32) {
        // shared B fragment (W^T row, contiguous in K)
        v8bf blo = *(const v8bf*)(wrow + kc + 8 * h);
        v8bf bhi = *(const v8bf*)(wrow + kc + 16 + 8 * h);
        v16bf bb;
#pragma unroll
        for (int t = 0; t < 8; ++t) { bb[t] = blo[t]; bb[t + 8] = bhi[t]; }
#pragma unroll
        for (int mt = 0; mt < 4; ++mt) {
            v16bf a;
            if (!transA) {
                const float* ar = Ab + (size_t)((ti0 + mt) * 16 + ln) * L_;
                v4f f0 = *(const v4f*)(ar + kc + 8 * h);
                v4f f1 = *(const v4f*)(ar + kc + 8 * h + 4);
                v4f f2 = *(const v4f*)(ar + kc + 16 + 8 * h);
                v4f f3 = *(const v4f*)(ar + kc + 16 + 8 * h + 4);
#pragma unroll
                for (int t = 0; t < 4; ++t) {
                    a[t]      = (bf16_t)f0[t];
                    a[t + 4]  = (bf16_t)f1[t];
                    a[t + 8]  = (bf16_t)f2[t];
                    a[t + 12] = (bf16_t)f3[t];
                }
            } else {
                int col = (ti0 + mt) * 16 + ln;     // A^T row = A column
#pragma unroll
                for (int t = 0; t < 8; ++t) {
                    a[t]     = (bf16_t)Ab[(size_t)(kc + 8 * h + t) * L_ + col];
                    a[t + 8] = (bf16_t)Ab[(size_t)(kc + 16 + 8 * h + t) * L_ + col];
                }
            }
            c[mt] = __builtin_amdgcn_wmma_f32_16x16x32_bf16(false, a, false, bb,
                                                            (short)0, c[mt], false, false);
        }
    }

    int n = tn * 16 + ln;
    if (n < D_) {
#pragma unroll
        for (int mt = 0; mt < 4; ++mt) {
#pragma unroll
            for (int r = 0; r < 8; ++r) {
                int m = (ti0 + mt) * 16 + r + 8 * h;
                F[((size_t)b * L_ + m) * D_ + n] = c[mt][r];
            }
        }
    }
}

// ---------------------------------------------------------------------------
// W^T pack: Wt[layer][d][j] = bf16(Ws[layer][j][d]) (zero pad d in [D_,NP_))
// ---------------------------------------------------------------------------
__global__ void wtrans_kernel(const float* __restrict__ Ws, bf16_t* __restrict__ Wt) {
    int idx = blockIdx.x * blockDim.x + threadIdx.x;
    if (idx >= NLAYER_ * NP_ * L_) return;
    int layer = idx / (NP_ * L_);
    int rem   = idx - layer * NP_ * L_;
    int d = rem / L_;
    int j = rem - d * L_;
    float v = (d < D_) ? Ws[((size_t)layer * L_ + j) * D_ + d] : 0.0f;
    Wt[idx] = (bf16_t)v;
}

// ---------------------------------------------------------------------------
// Conv2d (2->1 ch, 3x3, same pad) + tanh.  kern = 18 floats, bias = 1 float.
// ---------------------------------------------------------------------------
__global__ void conv_tanh_kernel(const float* __restrict__ e, const float* __restrict__ f,
                                 const float* __restrict__ kern, const float* __restrict__ bias,
                                 float* __restrict__ o) {
    __shared__ float k[18];
    __shared__ float bs;
    if (threadIdx.x < 18) k[threadIdx.x] = kern[threadIdx.x];
    if (threadIdx.x == 0) bs = bias[0];
    __syncthreads();
    int idx = blockIdx.x * blockDim.x + threadIdx.x;
    if (idx >= B_ * L_ * D_) return;
    int d   = idx % D_;
    int bl  = idx / D_;
    int l   = bl % L_;
    int b   = bl / L_;
    float acc = bs;
#pragma unroll
    for (int ch = 0; ch < 2; ++ch) {
        const float* src = ch ? f : e;
#pragma unroll
        for (int dy = -1; dy <= 1; ++dy) {
            int ll = l + dy;
            if (ll < 0 || ll >= L_) continue;
#pragma unroll
            for (int dx = -1; dx <= 1; ++dx) {
                int dd = d + dx;
                if (dd < 0 || dd >= D_) continue;
                acc += src[((size_t)b * L_ + ll) * D_ + dd] * k[ch * 9 + (dy + 1) * 3 + (dx + 1)];
            }
        }
    }
    o[idx] = tanhf(acc);
}

// ---------------------------------------------------------------------------
// Row sums (w1[b,i] = sum_j A) and column sums (w2[b,j] = sum_i A).
// ---------------------------------------------------------------------------
__global__ void rowcolsum_kernel(const float* __restrict__ A,
                                 float* __restrict__ w1, float* __restrict__ w2) {
    int idx = blockIdx.x * blockDim.x + threadIdx.x;
    if (idx >= 2 * B_ * L_) return;
    int which = idx / (B_ * L_);
    int r = idx - which * (B_ * L_);
    int b = r / L_;
    int i = r - b * L_;
    const float* Ab = A + (size_t)b * L_ * L_;
    float s = 0.0f;
    if (!which) { for (int j = 0; j < L_; ++j) s += Ab[(size_t)i * L_ + j]; w1[r] = s; }
    else        { for (int j = 0; j < L_; ++j) s += Ab[(size_t)j * L_ + i]; w2[r] = s; }
}

// ---------------------------------------------------------------------------
// e += avg_pool3(o * w) along L (count_include_pad, /3)
// ---------------------------------------------------------------------------
__global__ void pool_update_kernel(const float* __restrict__ o, const float* __restrict__ w,
                                   float* __restrict__ e) {
    int idx = blockIdx.x * blockDim.x + threadIdx.x;
    if (idx >= B_ * L_ * D_) return;
    int d  = idx % D_;
    int bl = idx / D_;
    int l  = bl % L_;
    int b  = bl / L_;
    const float* ob = o + ((size_t)b * L_) * D_ + d;
    const float* wb = w + (size_t)b * L_;
    float t0 = (l > 0)      ? ob[(size_t)(l - 1) * D_] * wb[l - 1] : 0.0f;
    float t1 =                ob[(size_t)l * D_]       * wb[l];
    float t2 = (l < L_ - 1) ? ob[(size_t)(l + 1) * D_] * wb[l + 1] : 0.0f;
    e[idx] += (t0 + t1 + t2) * (1.0f / 3.0f);
}

// ---------------------------------------------------------------------------
// Head: h = x@fc1 + b; LayerNorm; ReLU; out = h@fc2 + b.  One block per batch.
// ---------------------------------------------------------------------------
__global__ __launch_bounds__(LIN_)
void head_kernel(const float* __restrict__ x,
                 const float* __restrict__ fc1w, const float* __restrict__ fc1b,
                 const float* __restrict__ lng,  const float* __restrict__ lnb,
                 const float* __restrict__ fc2w, const float* __restrict__ fc2b,
                 float* __restrict__ out) {
    int b = blockIdx.x;
    int j = threadIdx.x;
    __shared__ float xs[FIN_];
    __shared__ float red[LIN_];
    for (int t = j; t < FIN_; t += LIN_) xs[t] = x[(size_t)b * FIN_ + t];
    __syncthreads();
    float h = fc1b[j];
    for (int k = 0; k < FIN_; ++k) h += xs[k] * fc1w[(size_t)k * LIN_ + j];
    // mean
    red[j] = h; __syncthreads();
    for (int s = LIN_ / 2; s > 0; s >>= 1) { if (j < s) red[j] += red[j + s]; __syncthreads(); }
    float mu = red[0] * (1.0f / LIN_); __syncthreads();
    float dm = h - mu;
    red[j] = dm * dm; __syncthreads();
    for (int s = LIN_ / 2; s > 0; s >>= 1) { if (j < s) red[j] += red[j + s]; __syncthreads(); }
    float var = red[0] * (1.0f / LIN_); __syncthreads();
    h = dm * rsqrtf(var + EPS_) * lng[j] + lnb[j];
    h = fmaxf(h, 0.0f);
    // fc2 column 0
    red[j] = h * fc2w[(size_t)j * 2 + 0]; __syncthreads();
    for (int s = LIN_ / 2; s > 0; s >>= 1) { if (j < s) red[j] += red[j + s]; __syncthreads(); }
    if (j == 0) out[(size_t)b * 2 + 0] = red[0] + fc2b[0];
    __syncthreads();
    // fc2 column 1
    red[j] = h * fc2w[(size_t)j * 2 + 1]; __syncthreads();
    for (int s = LIN_ / 2; s > 0; s >>= 1) { if (j < s) red[j] += red[j + s]; __syncthreads(); }
    if (j == 0) out[(size_t)b * 2 + 1] = red[0] + fc2b[1];
}

// ---------------------------------------------------------------------------
extern "C" void kernel_launch(void* const* d_in, const int* in_sizes, int n_in,
                              void* d_out, int out_size, void* d_ws, size_t ws_size,
                              hipStream_t stream) {
    const int*   q1    = (const int*)d_in[0];
    const int*   q2    = (const int*)d_in[1];
    const float* emb   = (const float*)d_in[2];
    const float* Ws    = (const float*)d_in[3];
    const float* convk = (const float*)d_in[4];
    const float* convb = (const float*)d_in[5];
    const float* fc1w  = (const float*)d_in[6];
    const float* fc1b  = (const float*)d_in[7];
    const float* lng   = (const float*)d_in[8];
    const float* lnb   = (const float*)d_in[9];
    const float* fc2w  = (const float*)d_in[10];
    const float* fc2b  = (const float*)d_in[11];
    float* out = (float*)d_out;

    // workspace layout (256B aligned)
    char* ws = (char*)d_ws;
    size_t off = 0;
    auto alloc = [&](size_t bytes) -> char* {
        char* p = ws + off;
        off += (bytes + 255) & ~(size_t)255;
        return p;
    };
    const size_t EBYTES  = (size_t)B_ * L_ * D_ * sizeof(float);
    const size_t SBBYTES = (size_t)B_ * L_ * KP_ * sizeof(bf16_t);
    const size_t ABYTES  = (size_t)B_ * L_ * L_ * sizeof(float);
    const size_t VBYTES  = (size_t)B_ * L_ * sizeof(float);

    float*  e1  = (float*)alloc(EBYTES);
    float*  e2  = (float*)alloc(EBYTES);
    float*  o1  = (float*)alloc(EBYTES);
    float*  o2  = (float*)alloc(EBYTES);
    float*  f1  = (float*)alloc(EBYTES);
    float*  f2  = (float*)alloc(EBYTES);
    bf16_t* sb1 = (bf16_t*)alloc(SBBYTES);
    bf16_t* sb2 = (bf16_t*)alloc(SBBYTES);
    float*  A   = (float*)alloc(ABYTES);
    float*  sq1 = (float*)alloc(VBYTES);
    float*  sq2 = (float*)alloc(VBYTES);
    float*  v1  = (float*)alloc(VBYTES);
    float*  v2  = (float*)alloc(VBYTES);
    float*  w1  = (float*)alloc(VBYTES);
    float*  w2  = (float*)alloc(VBYTES);
    bf16_t* Wt  = (bf16_t*)alloc((size_t)NLAYER_ * NP_ * L_ * sizeof(bf16_t));
    float*  x   = (float*)alloc((size_t)B_ * FIN_ * sizeof(float));
    (void)ws_size; (void)in_sizes; (void)n_in; (void)out_size;

    const int nEl = B_ * L_ * D_;
    const dim3 blk256(256);

    gather_kernel<<<(nEl + 255) / 256, blk256, 0, stream>>>(q1, emb, e1, v1);
    gather_kernel<<<(nEl + 255) / 256, blk256, 0, stream>>>(q2, emb, e2, v2);
    wtrans_kernel<<<(NLAYER_ * NP_ * L_ + 255) / 256, blk256, 0, stream>>>(Ws, Wt);
    mean_kernel<<<(B_ * D_ + 255) / 256, blk256, 0, stream>>>(e1, x, 0);
    mean_kernel<<<(B_ * D_ + 255) / 256, blk256, 0, stream>>>(e2, x, 3 * D_);

    for (int layer = 0; layer < NLAYER_; ++layer) {
        const bf16_t* Wtl = Wt + (size_t)layer * NP_ * L_;
        const float*  kl  = convk + (size_t)layer * 18;
        const float*  bl  = convb + (size_t)layer;

        // Wide_Conv: A = match_score(e1,e2); f = A@W, A^T@W; o = tanh(conv)
        pack_kernel<<<B_ * L_, 128, 0, stream>>>(e1, v1, sb1, sq1);
        pack_kernel<<<B_ * L_, 128, 0, stream>>>(e2, v2, sb2, sq2);
        msgemm_kernel<<<dim3(L_ / 16, L_ / 64, B_), 32, 0, stream>>>(sb1, sb2, sq1, sq2, A);
        fgemm_kernel<<<dim3(L_ / 64, NP_ / 16, B_), 32, 0, stream>>>(A, Wtl, f1, 0);
        fgemm_kernel<<<dim3(L_ / 64, NP_ / 16, B_), 32, 0, stream>>>(A, Wtl, f2, 1);
        conv_tanh_kernel<<<(nEl + 255) / 256, blk256, 0, stream>>>(e1, f1, kl, bl, o1);
        conv_tanh_kernel<<<(nEl + 255) / 256, blk256, 0, stream>>>(e2, f2, kl, bl, o2);
        mean_kernel<<<(B_ * D_ + 255) / 256, blk256, 0, stream>>>(o1, x, (1 + layer) * D_);
        mean_kernel<<<(B_ * D_ + 255) / 256, blk256, 0, stream>>>(o2, x, 3 * D_ + (1 + layer) * D_);

        // attention_avg_pooling: A2 = match_score(o1,o2); e += avgpool3(o*w)
        pack_kernel<<<B_ * L_, 128, 0, stream>>>(o1, v1, sb1, sq1);
        pack_kernel<<<B_ * L_, 128, 0, stream>>>(o2, v2, sb2, sq2);
        msgemm_kernel<<<dim3(L_ / 16, L_ / 64, B_), 32, 0, stream>>>(sb1, sb2, sq1, sq2, A);
        rowcolsum_kernel<<<(2 * B_ * L_ + 255) / 256, blk256, 0, stream>>>(A, w1, w2);
        pool_update_kernel<<<(nEl + 255) / 256, blk256, 0, stream>>>(o1, w1, e1);
        pool_update_kernel<<<(nEl + 255) / 256, blk256, 0, stream>>>(o2, w2, e2);
    }

    head_kernel<<<B_, LIN_, 0, stream>>>(x, fc1w, fc1b, lng, lnb, fc2w, fc2b, out);
}
